// DualStreamPhasorBlock_37082747633853
// MI455X (gfx1250) — matrix operand order repaired and checked by reading, not compiled, not run on hardware
//
#include <hip/hip_runtime.h>
#include <hip/hip_bf16.h>

typedef __attribute__((ext_vector_type(16))) _Float16 v16h;
typedef __attribute__((ext_vector_type(8)))  _Float16 v8h;
typedef __attribute__((ext_vector_type(8)))  float    v8f;

#define PI_F 3.14159265358979323846f

// ---------------- elementwise f32 -> f16 ----------------
__global__ void k_f32_to_f16(const float* __restrict__ in, _Float16* __restrict__ out, int n) {
  int i = blockIdx.x * blockDim.x + threadIdx.x;
  if (i < n) out[i] = (_Float16)in[i];
}

// ---------------- transpose+convert: W (R x C, f32) -> WT (C x R, f16) ----------------
__global__ void k_transpose_f16(const float* __restrict__ W, _Float16* __restrict__ WT, int R, int C) {
  int i = blockIdx.x * blockDim.x + threadIdx.x;
  if (i < R * C) {
    int r = i / C, c = i % C;
    WT[c * R + r] = (_Float16)W[i];
  }
}

// ---------------- WMMA fragment loads (CDNA5 16x16x32 f16 layouts, ISA 7.12.2) ----------------
// A (16x32, row-major, stride lda): lane m=l&15, h=l>>4 holds K = h*8+{0..7} and 16+h*8+{0..7}
__device__ inline v16h frag_a(const _Float16* __restrict__ A, int lda, int k0, int lane) {
  int m = lane & 15, h = lane >> 4;
  const _Float16* p = A + m * lda + k0;
  v8h lo = *(const v8h*)(p + h * 8);
  v8h hi = *(const v8h*)(p + 16 + h * 8);
  v16h r;
#pragma unroll
  for (int i = 0; i < 8; ++i) { r[i] = lo[i]; r[i + 8] = hi[i]; }
  return r;
}
// B (32x16) from BT stored N x Kdim row-major: lane n=l&15, h=l>>4 holds K = h*8+{0..7} and 16+h*8+{0..7} of column n
__device__ inline v16h frag_b(const _Float16* __restrict__ BT, int ldb, int k0, int lane) {
  int n = lane & 15, h = lane >> 4;
  const _Float16* p = BT + n * ldb + k0;
  v8h lo = *(const v8h*)(p + h * 8);
  v8h hi = *(const v8h*)(p + 16 + h * 8);
  v16h r;
#pragma unroll
  for (int i = 0; i < 8; ++i) { r[i] = lo[i]; r[i + 8] = hi[i]; }
  return r;
}

// ---------------- generic wave-level WMMA GEMM ----------------
// One wave computes a 16x16 tile of C = A(Mx Kdim) * B(Kdim x N) + bias, with epilogue:
// act 0: Cf = val (+ residual)          (f32 out)
// act 1: Ch = tanh(val)                 (f16 out)
// act 2: Ch = relu(val)                 (f16 out)
// act 3: Pc = cos(PI*tanh(val)), Ps = sin(PI*tanh(val))   (two f32 outs)
__global__ void k_wmma_gemm(const _Float16* __restrict__ A, int lda,
                            const _Float16* __restrict__ BT, int ldb,
                            const float* __restrict__ bias,
                            float* __restrict__ Cf, _Float16* __restrict__ Ch,
                            float* __restrict__ Pc, float* __restrict__ Ps,
                            const float* __restrict__ residual,
                            int ldc, int Kdim, int act) {
  int lane = threadIdx.x;
  int tm = blockIdx.x * 16, tn = blockIdx.y * 16;
  v8f acc = {};
  for (int k0 = 0; k0 < Kdim; k0 += 32) {
    v16h a = frag_a(A + tm * lda, lda, k0, lane);
    v16h b = frag_b(BT + tn * ldb, ldb, k0, lane);
    acc = __builtin_amdgcn_wmma_f32_16x16x32_f16(false, a, false, b, (short)0, acc, false, false);
  }
  int n = tn + (lane & 15);
  int m0 = tm + 8 * (lane >> 4);
  float bv = bias ? bias[n] : 0.0f;
#pragma unroll
  for (int v = 0; v < 8; ++v) {
    int m = m0 + v;
    float val = acc[v] + bv;
    if (act == 0) {
      float r = residual ? residual[m * ldc + n] : 0.0f;
      Cf[m * ldc + n] = val + r;
    } else if (act == 1) {
      Ch[m * ldc + n] = (_Float16)tanhf(val);
    } else if (act == 2) {
      Ch[m * ldc + n] = (_Float16)fmaxf(val, 0.0f);
    } else {
      float ph = PI_F * tanhf(val);
      Pc[m * ldc + n] = cosf(ph);
      Ps[m * ldc + n] = sinf(ph);
    }
  }
}

// ---------------- gate logits + softmax (N=2) ----------------
__global__ void k_gates(const _Float16* __restrict__ g1, const float* __restrict__ Wg2,
                        const float* __restrict__ bg2, float* __restrict__ gates, int M) {
  int t = blockIdx.x * blockDim.x + threadIdx.x;
  if (t >= M) return;
  const _Float16* g = g1 + t * 64;
  float a0 = bg2[0], a1 = bg2[1];
  for (int j = 0; j < 64; ++j) {
    float gv = (float)g[j];
    a0 += gv * Wg2[j * 2 + 0];
    a1 += gv * Wg2[j * 2 + 1];
  }
  float mx = fmaxf(a0, a1);
  float e0 = expf(a0 - mx), e1 = expf(a1 - mx);
  float inv = 1.0f / (e0 + e1);
  gates[t * 2 + 0] = e0 * inv;
  gates[t * 2 + 1] = e1 * inv;
}

// ---------------- positional stream: per-(b,d) serial scan over L ----------------
__global__ void k_pos_scan(const float* __restrict__ base_phases, const float* __restrict__ vp,
                           float* __restrict__ pos_ret, int L, int D) {
  int b = blockIdx.x;
  int d = threadIdx.x;
  float re = 0.0f, im = 0.0f;
  for (int t = 0; t < L; ++t) {
    float ph = base_phases[t * D + d];
    float c = cosf(ph), s = sinf(ph);
    float v = vp[(b * L + t) * D + d];
    re += v * c;
    im += v * s;
    pos_ret[(b * L + t) * D + d] = (re * c + im * s) * rsqrtf((float)(t + 1));
  }
}

// ---------------- content stream: block per batch, thread per d, K-state in registers ----------------
#define TS 16
__global__ void k_content_scan(const float* __restrict__ cosk, const float* __restrict__ sink,
                               const float* __restrict__ cosq, const float* __restrict__ sinq,
                               const float* __restrict__ vc, float* __restrict__ cont_ret,
                               int L, int D) {
  __shared__ float sck[TS * 32], ssk[TS * 32], scq[TS * 32], ssq[TS * 32];
  int b = blockIdx.x;
  int d = threadIdx.x;
  float mr[32], mi[32];
#pragma unroll
  for (int k = 0; k < 32; ++k) { mr[k] = 0.0f; mi[k] = 0.0f; }
  for (int t0 = 0; t0 < L; t0 += TS) {
    __syncthreads();
    for (int idx = threadIdx.x; idx < TS * 32; idx += blockDim.x) {
      int g = (b * L + t0) * 32 + idx;   // rows of (B,L,32) are contiguous
      sck[idx] = cosk[g]; ssk[idx] = sink[g];
      scq[idx] = cosq[g]; ssq[idx] = sinq[g];
    }
    __syncthreads();
    for (int tt = 0; tt < TS; ++tt) {
      int t = t0 + tt;
      float v = vc[(b * L + t) * D + d];
      float acc = 0.0f;
#pragma unroll
      for (int k = 0; k < 32; ++k) {
        mr[k] += sck[tt * 32 + k] * v;
        mi[k] += ssk[tt * 32 + k] * v;
        acc += mr[k] * scq[tt * 32 + k] + mi[k] * ssq[tt * 32 + k];
      }
      cont_ret[(b * L + t) * D + d] = acc * rsqrtf((float)((t + 1) * 32));
    }
  }
}

// ---------------- gate-combine + layernorm -> f16 (block per token, 256 threads) ----------------
__global__ void k_combine_ln(const float* __restrict__ pos_ret, const float* __restrict__ cont_ret,
                             const float* __restrict__ gates, const float* __restrict__ ln_g,
                             const float* __restrict__ ln_b, _Float16* __restrict__ normed_h, int D) {
  __shared__ float red[256];
  int t = blockIdx.x;
  int d = threadIdx.x;
  float g0 = gates[t * 2 + 0], g1 = gates[t * 2 + 1];
  float c = g0 * pos_ret[t * D + d] + g1 * cont_ret[t * D + d];
  red[d] = c;
  __syncthreads();
  for (int s = 128; s > 0; s >>= 1) { if (d < s) red[d] += red[d + s]; __syncthreads(); }
  float mu = red[0] / (float)D;
  __syncthreads();
  float diff = c - mu;
  red[d] = diff * diff;
  __syncthreads();
  for (int s = 128; s > 0; s >>= 1) { if (d < s) red[d] += red[d + s]; __syncthreads(); }
  float var = red[0] / (float)D;
  float nv = diff * rsqrtf(var + 1e-5f) * ln_g[d] + ln_b[d];
  normed_h[t * D + d] = (_Float16)nv;
}

extern "C" void kernel_launch(void* const* d_in, const int* in_sizes, int n_in,
                              void* d_out, int out_size, void* d_ws, size_t ws_size,
                              hipStream_t stream) {
  (void)in_sizes; (void)n_in; (void)out_size; (void)ws_size;
  const int D = 256, K = 32, Bb = 2, L = 2048;
  const int M = Bb * L;  // 4096 tokens

  const float* x           = (const float*)d_in[0];
  const float* base_phases = (const float*)d_in[1];
  const float* Wk1 = (const float*)d_in[2];  const float* bk1 = (const float*)d_in[3];
  const float* Wk2 = (const float*)d_in[4];  const float* bk2 = (const float*)d_in[5];
  const float* Wq1 = (const float*)d_in[6];  const float* bq1 = (const float*)d_in[7];
  const float* Wq2 = (const float*)d_in[8];  const float* bq2 = (const float*)d_in[9];
  const float* Wvc = (const float*)d_in[10]; const float* bvc = (const float*)d_in[11];
  const float* Wvp = (const float*)d_in[12]; const float* bvp = (const float*)d_in[13];
  const float* Wg1 = (const float*)d_in[14]; const float* bg1 = (const float*)d_in[15];
  const float* Wg2 = (const float*)d_in[16]; const float* bg2 = (const float*)d_in[17];
  const float* ln_g = (const float*)d_in[18]; const float* ln_b = (const float*)d_in[19];
  const float* Wo  = (const float*)d_in[20]; const float* bo  = (const float*)d_in[21];
  float* out = (float*)d_out;

  // workspace carve-up (256B aligned)
  char* ws = (char*)d_ws;
  size_t off = 0;
  auto alloc = [&](size_t bytes) -> void* {
    void* p = ws + off;
    off += (bytes + 255) & ~(size_t)255;
    return p;
  };
  _Float16* xh     = (_Float16*)alloc((size_t)M * D * 2);
  _Float16* WTk1   = (_Float16*)alloc((size_t)D * D * 2);
  _Float16* WTq1   = (_Float16*)alloc((size_t)D * D * 2);
  _Float16* WTvc   = (_Float16*)alloc((size_t)D * D * 2);
  _Float16* WTvp   = (_Float16*)alloc((size_t)D * D * 2);
  _Float16* WTo    = (_Float16*)alloc((size_t)D * D * 2);
  _Float16* WTk2   = (_Float16*)alloc((size_t)K * D * 2);
  _Float16* WTq2   = (_Float16*)alloc((size_t)K * D * 2);
  _Float16* WTg1   = (_Float16*)alloc((size_t)64 * D * 2);
  _Float16* hk_h   = (_Float16*)alloc((size_t)M * D * 2);
  _Float16* hq_h   = (_Float16*)alloc((size_t)M * D * 2);
  _Float16* g1h    = (_Float16*)alloc((size_t)M * 64 * 2);
  _Float16* norm_h = (_Float16*)alloc((size_t)M * D * 2);
  float* cosk = (float*)alloc((size_t)M * K * 4);
  float* sink = (float*)alloc((size_t)M * K * 4);
  float* cosq = (float*)alloc((size_t)M * K * 4);
  float* sinq = (float*)alloc((size_t)M * K * 4);
  float* vcb  = (float*)alloc((size_t)M * D * 4);
  float* vpb  = (float*)alloc((size_t)M * D * 4);
  float* posr = (float*)alloc((size_t)M * D * 4);
  float* conr = (float*)alloc((size_t)M * D * 4);
  float* gts  = (float*)alloc((size_t)M * 2 * 4);

  // 1) precision conversion / weight transpose
  {
    int n = M * D;
    k_f32_to_f16<<<(n + 255) / 256, 256, 0, stream>>>(x, xh, n);
    int nw = D * D;
    k_transpose_f16<<<(nw + 255) / 256, 256, 0, stream>>>(Wk1, WTk1, D, D);
    k_transpose_f16<<<(nw + 255) / 256, 256, 0, stream>>>(Wq1, WTq1, D, D);
    k_transpose_f16<<<(nw + 255) / 256, 256, 0, stream>>>(Wvc, WTvc, D, D);
    k_transpose_f16<<<(nw + 255) / 256, 256, 0, stream>>>(Wvp, WTvp, D, D);
    k_transpose_f16<<<(nw + 255) / 256, 256, 0, stream>>>(Wo,  WTo,  D, D);
    int nk = D * K;
    k_transpose_f16<<<(nk + 255) / 256, 256, 0, stream>>>(Wk2, WTk2, D, K);
    k_transpose_f16<<<(nk + 255) / 256, 256, 0, stream>>>(Wq2, WTq2, D, K);
    int ng = D * 64;
    k_transpose_f16<<<(ng + 255) / 256, 256, 0, stream>>>(Wg1, WTg1, D, 64);
  }

  dim3 blk(32, 1, 1);
  dim3 gDD(M / 16, D / 16);   // 256 x 16
  dim3 gDK(M / 16, K / 16);   // 256 x 2
  dim3 gDG(M / 16, 64 / 16);  // 256 x 4

  // 2) token-parallel WMMA GEMMs
  // hk = tanh(x@Wk1+bk1);  key phases -> cos/sin
  k_wmma_gemm<<<gDD, blk, 0, stream>>>(xh, D, WTk1, D, bk1, nullptr, hk_h, nullptr, nullptr, nullptr, D, D, 1);
  k_wmma_gemm<<<gDK, blk, 0, stream>>>(hk_h, D, WTk2, D, bk2, nullptr, nullptr, cosk, sink, nullptr, K, D, 3);
  // hq = tanh(x@Wq1+bq1);  query phases -> cos/sin
  k_wmma_gemm<<<gDD, blk, 0, stream>>>(xh, D, WTq1, D, bq1, nullptr, hq_h, nullptr, nullptr, nullptr, D, D, 1);
  k_wmma_gemm<<<gDK, blk, 0, stream>>>(hq_h, D, WTq2, D, bq2, nullptr, nullptr, cosq, sinq, nullptr, K, D, 3);
  // value streams
  k_wmma_gemm<<<gDD, blk, 0, stream>>>(xh, D, WTvc, D, bvc, vcb, nullptr, nullptr, nullptr, nullptr, D, D, 0);
  k_wmma_gemm<<<gDD, blk, 0, stream>>>(xh, D, WTvp, D, bvp, vpb, nullptr, nullptr, nullptr, nullptr, D, D, 0);
  // gate hidden
  k_wmma_gemm<<<gDG, blk, 0, stream>>>(xh, D, WTg1, D, bg1, nullptr, g1h, nullptr, nullptr, nullptr, 64, D, 2);

  // 3) gates softmax
  k_gates<<<(M + 255) / 256, 256, 0, stream>>>(g1h, Wg2, bg2, gts, M);

  // 4) scans (sequential over L)
  k_pos_scan<<<Bb, D, 0, stream>>>(base_phases, vpb, posr, L, D);
  k_content_scan<<<Bb, D, 0, stream>>>(cosk, sink, cosq, sinq, vcb, conr, L, D);

  // 5) combine + layernorm -> f16
  k_combine_ln<<<M, D, 0, stream>>>(posr, conr, gts, ln_g, ln_b, norm_h, D);

  // 6) out = x + normed @ Wo + bo   (WMMA with fused residual)
  k_wmma_gemm<<<gDD, blk, 0, stream>>>(norm_h, D, WTo, D, bo, out, nullptr, nullptr, nullptr, x, D, D, 0);
}
